// LightningAttention_11879879541691
// MI455X (gfx1250) — compile-verified, matrix operand order checked by reading, not run on hardware
//
#include <hip/hip_runtime.h>

// ---------------------------------------------------------------------------
// Flash attention, B=4, S=4096, D=512, fp32 in/out, gfx1250 (MI455X).
// Compute-bound: both GEMMs on v_wmma_f32_16x16x32_f16 (f32 accum), online
// softmax, single pass. 8 waves/WG, 32 q-rows/WG, 64-key tiles, split-K score
// reduction across 4 d-slice waves. K/V/P/partials stored in LDS in WMMA-
// fragment-packed order (fragment load = 2x ds_load_b128); fragment batches
// are double-buffered so LDS latency overlaps the WMMA pipe. Mask tile is
// copied global->LDS with GLOBAL_LOAD_ASYNC_TO_LDS_B128 + s_wait_asynccnt.
// ---------------------------------------------------------------------------

typedef _Float16 half_t;
typedef __attribute__((ext_vector_type(16))) _Float16 v16h;
typedef __attribute__((ext_vector_type(8)))  _Float16 v8h;
typedef __attribute__((ext_vector_type(8)))  float    v8f;
typedef __attribute__((ext_vector_type(4)))  int      v4i;

#define S_LEN   4096
#define D_DIM   512
#define NBATCH  4
#define BQ      32          // query rows per workgroup
#define BC      64          // keys per tile
#define DSLICE  128         // output-D columns per wave
#define NWAVE   8
#define NTILES  (S_LEN / BC)

#if defined(__gfx1250__) && __has_builtin(__builtin_amdgcn_global_load_async_to_lds_b128)
#define USE_ASYNC_LDS 1
#else
#define USE_ASYNC_LDS 0
#endif

// Builtin wants typed, address-space-qualified pointers:
//   arg0: int4 addrspace(1)* (global source), arg1: int4 addrspace(3)* (LDS dest)
typedef __attribute__((address_space(1))) v4i g_v4i;
typedef __attribute__((address_space(3))) v4i l_v4i;
__device__ __forceinline__ g_v4i* as_glob_v4i(const void* p) {
    return (g_v4i*)(uintptr_t)p;
}
__device__ __forceinline__ l_v4i* as_lds_v4i(void* p) {
    return (l_v4i*)(uint32_t)(uintptr_t)p;   // low 32 bits of generic = LDS offset
}

// Element e of a v16h fragment maps to K = 16*(e>=8) + 8*hs + (e&7)  (ISA 7.12.2)
__device__ __forceinline__ int kmap(int e, int hs) {
    return ((e >> 3) << 4) + (hs << 3) + (e & 7);
}

__global__ __launch_bounds__(256, 1)
void fa_fwd_wmma4_gfx1250(const float* __restrict__ Q,
                          const float* __restrict__ K,
                          const float* __restrict__ V,
                          const float* __restrict__ M,
                          float* __restrict__ O) {
    extern __shared__ char smem_raw[];
    // sKt: K^T fragments: [f=d/32: 16][nt=key/16: 4][lane: 32][e: 16] f16 = 64 KB
    half_t* sKt   = (half_t*)smem_raw;
    // sV : V fragments:  [ks=key/32: 2][ntg=d/16: 32][lane: 32][e: 16] f16 = 64 KB
    half_t* sV    = sKt + D_DIM * BC;
    // sPart: score partials, C-frag packed: [wid: 8][nt: 4][lane: 32][r: 8] f32 = 32 KB
    float*  sPart = (float*)(sV + BC * D_DIM);
    // sP : P A-fragments: [wid: 8][f: 2][lane: 32][e: 16] f16 = 16 KB
    half_t* sP    = (half_t*)(sPart + NWAVE * 4 * 32 * 8);
    // sMask: [row: 32][col: 64] f32 = 8 KB
    float*  sMask = (float*)(sP + NWAVE * 2 * 32 * 16);

    const int tid  = threadIdx.x;
    const int wid  = tid >> 5;
    const int lane = tid & 31;
    const int hs   = lane >> 4;     // which 16-lane half
    const int ln   = lane & 15;
    const int qsub = wid >> 2;      // 0..1 : 16-row q sub-tile
    const int dsl  = wid & 3;       // 0..3 : 128-wide D slice
    const int dbase = dsl * DSLICE;

    const int b     = blockIdx.y;
    const int qwg   = blockIdx.x * BQ;
    const int qrow0 = qwg + qsub * 16;
    const size_t bOff = (size_t)b * S_LEN * D_DIM;
    const float scale = 0.044194173824159216f;   // 1/sqrt(512)

    // Resident Q A-fragments: rows qrow0..+15, cols dbase..+127 (4 frags of K=32)
    v16h qfrag[4];
    {
        const float* qp = Q + bOff + (size_t)(qrow0 + ln) * D_DIM + dbase;
        #pragma unroll
        for (int f = 0; f < 4; ++f)
            #pragma unroll
            for (int e = 0; e < 16; ++e)
                qfrag[f][e] = (half_t)qp[f * 32 + kmap(e, hs)];
    }

    v8f oacc[8];
    v8f mstate, lstate;
    #pragma unroll
    for (int r = 0; r < 8; ++r) { mstate[r] = -1e30f; lstate[r] = 0.0f; }
    #pragma unroll
    for (int o = 0; o < 8; ++o)
        #pragma unroll
        for (int r = 0; r < 8; ++r) oacc[o][r] = 0.0f;

    const v16h* sVv  = (const v16h*)sV;
    const v16h* sPv  = (const v16h*)sP;
    v8f*        sPartv = (v8f*)sPart;
    // per-wave fragment base pointers (all loop loads become base + imm offset)
    const v16h* kfragp = (const v16h*)sKt + (size_t)(dsl * 16) * 32 + lane;  // [ks*4+nt]*32
    const v16h* vfragp = sVv + (size_t)(dsl * 8) * 32 + lane;                // [ks*32+nt]*32
    const v16h* pfragp = sPv + (size_t)(wid * 2) * 32 + lane;                // [f]*32

    for (int kt = 0; kt < NTILES; ++kt) {
        const int kbase = kt * BC;
        __syncthreads();   // protect LDS tiles/partials/mask from previous iter

        // ---- mask tile: pure f32 copy -> async load to LDS ----
        {
            #pragma unroll
            for (int it = 0; it < 2; ++it) {
                const int idx = tid + it * 256;          // 512 float4
                const int row = idx >> 4;
                const int c4  = (idx & 15) << 2;
                const float* g = M + (size_t)(qwg + row) * S_LEN + kbase + c4;
                float* l = sMask + row * 64 + c4;
#if USE_ASYNC_LDS
                __builtin_amdgcn_global_load_async_to_lds_b128(as_glob_v4i(g), as_lds_v4i(l), 0, 0);
#else
                *(float4*)l = *(const float4*)g;
#endif
            }
        }

        // ---- K tile: row reads (2x b128), fragment-packed f16 store (1x b128) ----
        {
            const float* kp = K + bOff + (size_t)kbase * D_DIM;
            #pragma unroll
            for (int it = 0; it < 16; ++it) {
                const int idx = tid + it * 256;          // 4096 groups of 8
                const int key = idx >> 6;                // 64 d-groups per row
                const int d8  = (idx & 63) << 3;
                const float4 a = *(const float4*)(kp + key * D_DIM + d8);
                const float4 c = *(const float4*)(kp + key * D_DIM + d8 + 4);
                const int f      = d8 >> 5;
                const int klocal = d8 & 31;
                const int hsk    = (klocal >> 3) & 1;
                const int e0     = (klocal >> 4) << 3;
                const int lane_t = (key & 15) + (hsk << 4);
                const int nt     = key >> 4;
                v8h pk = { (half_t)a.x, (half_t)a.y, (half_t)a.z, (half_t)a.w,
                           (half_t)c.x, (half_t)c.y, (half_t)c.z, (half_t)c.w };
                *(v8h*)(sKt + (((f * 4 + nt) * 32 + lane_t) * 16 + e0)) = pk;
            }
        }

        // ---- V tile: column reads (8x b32, lane-coalesced), packed store ----
        {
            const float* vp = V + bOff + (size_t)kbase * D_DIM;
            #pragma unroll
            for (int it = 0; it < 16; ++it) {
                const int idx  = tid + it * 256;         // 4096 groups of 8
                const int kb   = idx >> 9;               // 8 key-groups of 8
                const int d    = idx & 511;
                const int key0 = kb << 3;
                v8h pk;
                #pragma unroll
                for (int j = 0; j < 8; ++j)
                    pk[j] = (half_t)vp[(key0 + j) * D_DIM + d];
                const int klocal = key0 & 31;
                const int hsv    = (klocal >> 3) & 1;
                const int e0     = (klocal >> 4) << 3;
                const int lane_t = (d & 15) + (hsv << 4);
                const int ntg    = d >> 4;
                const int ksf    = key0 >> 5;
                *(v8h*)(sV + (((ksf * 32 + ntg) * 32 + lane_t) * 16 + e0)) = pk;
            }
            if (kt + 1 < NTILES) {   // global_prefetch_b8 on next tile
                __builtin_prefetch(vp + BC * D_DIM + tid * 16, 0, 1);
                __builtin_prefetch(K + bOff + (size_t)(kbase + BC) * D_DIM + tid * 16, 0, 1);
            }
        }
#if USE_ASYNC_LDS
#if __has_builtin(__builtin_amdgcn_s_wait_asynccnt)
        __builtin_amdgcn_s_wait_asynccnt(0);
#else
        asm volatile("s_wait_asynccnt 0" ::: "memory");
#endif
#endif
        __syncthreads();

        // ---- partial scores over this wave's 128-wide d slice: 16 WMMAs ----
        // Double-buffer B-fragment batches: load slice ks+1 while ks computes.
        v8f sacc[4];
        #pragma unroll
        for (int nt = 0; nt < 4; ++nt)
            #pragma unroll
            for (int r = 0; r < 8; ++r) sacc[nt][r] = 0.0f;
        {
            v16h bcur[4], bnxt[4];
            #pragma unroll
            for (int nt = 0; nt < 4; ++nt) bcur[nt] = kfragp[nt * 32];
            #pragma unroll
            for (int ks = 0; ks < 4; ++ks) {
                if (ks < 3) {
                    #pragma unroll
                    for (int nt = 0; nt < 4; ++nt)
                        bnxt[nt] = kfragp[((ks + 1) * 4 + nt) * 32];
                }
                #pragma unroll
                for (int nt = 0; nt < 4; ++nt)
                    sacc[nt] = __builtin_amdgcn_wmma_f32_16x16x32_f16(
                        false, qfrag[ks], false, bcur[nt], (short)0, sacc[nt], false, false);
                #pragma unroll
                for (int nt = 0; nt < 4; ++nt) bcur[nt] = bnxt[nt];
            }
        }

        // ---- write partials (vector v8f stores), reduce across 4 waves ----
        #pragma unroll
        for (int nt = 0; nt < 4; ++nt)
            sPartv[((wid * 4 + nt) * 32) + lane] = sacc[nt];
        __syncthreads();

        v8f full[4];
        #pragma unroll
        for (int nt = 0; nt < 4; ++nt) {
            v8f a = sPartv[(((qsub * 4 + 0) * 4 + nt) * 32) + lane];
            a = a + sPartv[(((qsub * 4 + 1) * 4 + nt) * 32) + lane];
            a = a + sPartv[(((qsub * 4 + 2) * 4 + nt) * 32) + lane];
            a = a + sPartv[(((qsub * 4 + 3) * 4 + nt) * 32) + lane];
            #pragma unroll
            for (int r = 0; r < 8; ++r)
                full[nt][r] = a[r] * scale
                            + sMask[(qsub * 16 + r + hs * 8) * 64 + nt * 16 + ln];
        }

        // ---- online softmax (row = r + 8*hs, col = nt*16 + ln) ----
        v8f rowmax = full[0];
        #pragma unroll
        for (int nt = 1; nt < 4; ++nt)
            #pragma unroll
            for (int r = 0; r < 8; ++r) rowmax[r] = fmaxf(rowmax[r], full[nt][r]);
        #pragma unroll
        for (int sh = 8; sh >= 1; sh >>= 1)
            #pragma unroll
            for (int r = 0; r < 8; ++r)
                rowmax[r] = fmaxf(rowmax[r], __shfl_xor(rowmax[r], sh, 32));

        v8f mnew, alpha;
        #pragma unroll
        for (int r = 0; r < 8; ++r) {
            mnew[r]  = fmaxf(mstate[r], rowmax[r]);
            alpha[r] = __expf(mstate[r] - mnew[r]);
        }

        // P -> LDS in A-fragment-packed order (scatter b16; layout conversion)
        const int hs_t = (ln >> 3) & 1;
        v8f psum;
        #pragma unroll
        for (int r = 0; r < 8; ++r) psum[r] = 0.0f;
        #pragma unroll
        for (int nt = 0; nt < 4; ++nt) {
            const int f = nt >> 1;
            const int e = ((nt & 1) << 3) + (ln & 7);
            half_t* pb = sP + (((wid * 2 + f) * 32 + (hs_t << 4) + hs * 8) * 16) + e;
            #pragma unroll
            for (int r = 0; r < 8; ++r) {
                const float p = __expf(full[nt][r] - mnew[r]);
                psum[r] += p;
                pb[r * 16] = (half_t)p;
            }
        }

        // LDS is in-order within a wave: issue the P-fragment reads and the
        // first V-fragment batch now, and hide their latency under the
        // softmax-state / O-rescale VALU work below.
        v16h pfrag[2];
        #pragma unroll
        for (int f = 0; f < 2; ++f) pfrag[f] = pfragp[f * 32];
        v16h vcur[4], vnxt[4];
        #pragma unroll
        for (int j = 0; j < 4; ++j) vcur[j] = vfragp[j * 32];

        #pragma unroll
        for (int sh = 8; sh >= 1; sh >>= 1)
            #pragma unroll
            for (int r = 0; r < 8; ++r) psum[r] += __shfl_xor(psum[r], sh, 32);
        #pragma unroll
        for (int r = 0; r < 8; ++r) {
            lstate[r] = lstate[r] * alpha[r] + psum[r];
            mstate[r] = mnew[r];
        }
        #pragma unroll
        for (int o = 0; o < 8; ++o)
            #pragma unroll
            for (int r = 0; r < 8; ++r) oacc[o][r] *= alpha[r];

        // ---- O += P x V: 16 WMMAs, double-buffered 4-fragment batches ----
        #pragma unroll
        for (int g = 0; g < 4; ++g) {
            const int ks  = g >> 1;
            const int nt0 = (g & 1) * 4;
            if (g < 3) {
                const int gn  = g + 1;
                const int ksn = gn >> 1;
                const int ntn = (gn & 1) * 4;
                #pragma unroll
                for (int j = 0; j < 4; ++j)
                    vnxt[j] = vfragp[(ksn * 32 + ntn + j) * 32];
            }
            #pragma unroll
            for (int j = 0; j < 4; ++j)
                oacc[nt0 + j] = __builtin_amdgcn_wmma_f32_16x16x32_f16(
                    false, pfrag[ks], false, vcur[j], (short)0, oacc[nt0 + j], false, false);
            #pragma unroll
            for (int j = 0; j < 4; ++j) vcur[j] = vnxt[j];
        }
    }

    // ---- finalize: O / l, store fp32 ----
    float* op = O + bOff + (size_t)qrow0 * D_DIM + dbase;
    #pragma unroll
    for (int r = 0; r < 8; ++r) {
        const float inv = 1.0f / lstate[r];
        #pragma unroll
        for (int nt = 0; nt < 8; ++nt)
            op[(size_t)(r + hs * 8) * D_DIM + nt * 16 + ln] = oacc[nt][r] * inv;
    }
}

extern "C" void kernel_launch(void* const* d_in, const int* in_sizes, int n_in,
                              void* d_out, int out_size, void* d_ws, size_t ws_size,
                              hipStream_t stream) {
    const float* Q = (const float*)d_in[0];
    const float* K = (const float*)d_in[1];
    const float* V = (const float*)d_in[2];
    const float* M = (const float*)d_in[3];
    float* O = (float*)d_out;

    constexpr size_t SMEM = (size_t)D_DIM * BC * 2        // sKt   64 KB
                          + (size_t)BC * D_DIM * 2        // sV    64 KB
                          + (size_t)NWAVE * 4 * 32 * 8 * 4  // sPart 32 KB
                          + (size_t)NWAVE * 2 * 32 * 16 * 2 // sP    16 KB
                          + (size_t)32 * 64 * 4;            // sMask  8 KB  = 184 KB
    (void)hipFuncSetAttribute(reinterpret_cast<const void*>(&fa_fwd_wmma4_gfx1250),
                              hipFuncAttributeMaxDynamicSharedMemorySize, (int)SMEM);

    dim3 grid(S_LEN / BQ, NBATCH, 1);
    fa_fwd_wmma4_gfx1250<<<grid, 256, SMEM, stream>>>(Q, K, V, M, O);
}
